// RelativeSelfAttention_74328704024971
// MI455X (gfx1250) — compile-verified, matrix operand order checked by reading, not run on hardware
//
#include <hip/hip_runtime.h>
#include <hip/hip_bf16.h>

// ---------------------------------------------------------------------------
// RelativeSelfAttention for MI455X (gfx1250), bf16 WMMA pipeline.
//   B=4096 windows, C=49 tokens, N=512 channels, H=32 heads, D=16.
// 4 kernels: prep (weight transpose + padded bias), QKV GEMM, fused
// attention (per-head flash-style with softmax in registers), proj GEMM.
// Workspace requirement: ~825 MB (bf16 Q/K/V + attention output + weightsT).
//
// Round-1 fix: LICM was hoisting all LDS A-fragments out of the n-tile loop
// in the GEMM kernels (128 fragments = ~1024 VGPRs) and spilling them to
// scratch, so every WMMA read its A operand via scratch_load_b128. An
// asm-volatile memory clobber at the top of the n-tile loop keeps the
// ds_load_b128 fragment loads inside the loop and kills the spills.
// ---------------------------------------------------------------------------

typedef __attribute__((ext_vector_type(16))) __bf16 v16bf;
typedef __attribute__((ext_vector_type(8)))  float  v8f;

union FragAB {
  v16bf v;
  uint4 q[2];
};

__device__ __forceinline__ unsigned short f2bf(float f) {
  unsigned int x = __float_as_uint(f);
  x += 0x7FFFu + ((x >> 16) & 1u);         // round-to-nearest-even
  return (unsigned short)(x >> 16);
}

// bf16 fragment loader. Works for A (row-major M x K) and B (stored as N x K).
// ISA 7.12.2 (16-bit A 16x32): lanes 0-15 -> row lane, kbase 0; lanes 16-31 ->
// row lane-16, kbase 8. Element e: K = kbase + (e<8 ? e : 8+e)  -> two b128
// loads at half-offsets kbase and kbase+16.
__device__ __forceinline__ v16bf load_frag(const unsigned short* base, int row0,
                                           int rstride, int k0, int lane) {
  const int r  = row0 + (lane & 15);
  const int kb = k0 + ((lane >> 4) << 3);
  const unsigned short* p = base + r * rstride + kb;
  FragAB f;
  f.q[0] = *(const uint4*)(p);
  f.q[1] = *(const uint4*)(p + 16);
  return f.v;
}

// K=16 tile (head dim): real data in K 0..15 only, zero K 16..31.
__device__ __forceinline__ v16bf load_frag_k16(const unsigned short* base, int row0,
                                               int rstride, int lane) {
  const int r  = row0 + (lane & 15);
  const int kb = (lane >> 4) << 3;          // 0 or 8, both < 16
  FragAB f;
  f.q[0] = *(const uint4*)(base + r * rstride + kb);
  f.q[1] = make_uint4(0u, 0u, 0u, 0u);
  return f.v;
}

// ---------------------------------------------------------------------------
// K0: weight transposes (fp32 -> bf16, [N][K] layout) + padded bias table
//     biasPad[h][m][n] (64x64, -1e30 outside the 49x49 window).
// ---------------------------------------------------------------------------
__global__ __launch_bounds__(256) void k0_prep(const float* __restrict__ Wqkv,
                                               const float* __restrict__ Wproj,
                                               const float* __restrict__ btab,
                                               unsigned short* __restrict__ wqkvT,
                                               unsigned short* __restrict__ wprojT,
                                               float* __restrict__ biasPad) {
  int i = blockIdx.x * 256 + threadIdx.x;
  const int T1 = 1536 * 512, T2 = 512 * 512, T3 = 32 * 64 * 64;
  if (i < T1) {                                    // wqkvT[n][k]
    int n = i >> 9, k = i & 511;
    wqkvT[i] = f2bf(Wqkv[k * 1536 + n]);
  } else if ((i -= T1) < T2) {                     // wprojT[n][k]
    int n = i >> 9, k = i & 511;
    wprojT[i] = f2bf(Wproj[k * 512 + n]);
  } else if ((i -= T2) < T3) {                     // biasPad[h][m][n]
    int h = i >> 12, m = (i >> 6) & 63, n = i & 63;
    float v = -1e30f;
    if (m < 49 && n < 49) {
      int dh = (m / 7) - (n / 7) + 6;
      int dw = (m % 7) - (n % 7) + 6;
      v = btab[(dh * 13 + dw) * 32 + h];
    }
    biasPad[i] = v;
  }
}

// ---------------------------------------------------------------------------
// K1: qkv = x @ Wqkv + b.  M-block = 128 rows (1568 blocks exactly covers
// 200704 rows).  x staged in LDS as bf16 (row stride 520 halfs -> conflict
// free b128 reads).  B-fragments straight from L2-resident wqkvT.
// Each 16-col n-tile is exactly one (s,h) slice: col = s*512 + h*16 + d.
// Q is pre-scaled by 1/sqrt(32); Q/K/V written bf16 as [b][h][c][16].
// ---------------------------------------------------------------------------
__global__ __launch_bounds__(256) void k1_qkv(const float* __restrict__ x,
                                              const float* __restrict__ b_qkv,
                                              const unsigned short* __restrict__ wqkvT,
                                              unsigned short* __restrict__ Qg,
                                              unsigned short* __restrict__ Kg,
                                              unsigned short* __restrict__ Vg) {
  __shared__ __align__(16) unsigned short xs[128 * 520];
  const int tid = threadIdx.x;
  const size_t mbase = (size_t)blockIdx.x * 128;

  for (int i = tid; i < 128 * 128; i += 256) {     // stage + convert x
    int r = i >> 7, k4 = (i & 127) << 2;
    float4 v = *(const float4*)(x + (mbase + r) * 512 + k4);
    unsigned short* d = xs + r * 520 + k4;
    d[0] = f2bf(v.x); d[1] = f2bf(v.y); d[2] = f2bf(v.z); d[3] = f2bf(v.w);
  }
  __syncthreads();

  const int wave = tid >> 5, lane = tid & 31;
  const int n = lane & 15, hiM = (lane >> 4) << 3;
  const v8f vzero = {0.f,0.f,0.f,0.f,0.f,0.f,0.f,0.f};

  for (int j = 0; j < 12; ++j) {                   // 96 n-tiles / 8 waves
    // Block LICM from hoisting the per-iteration LDS fragment loads (they are
    // j-invariant; hoisting 128 fragments forced scratch spills in round 1).
    asm volatile("" ::: "memory");

    const int nt = wave + (j << 3);
    const int n0 = nt << 4;
    v8f acc[8];
#pragma unroll
    for (int mt = 0; mt < 8; ++mt) acc[mt] = vzero;

    for (int ks = 0; ks < 16; ++ks) {
      const v16bf bfrag = load_frag(wqkvT, n0, 512, ks << 5, lane);
#pragma unroll
      for (int mt = 0; mt < 8; ++mt) {
        const v16bf afrag = load_frag(xs, mt << 4, 520, ks << 5, lane);
        acc[mt] = __builtin_amdgcn_wmma_f32_16x16x32_bf16(
            false, afrag, false, bfrag, (short)0, acc[mt], false, false);
      }
    }

    const int s = n0 >> 9;
    const int h = (n0 >> 4) & 31;
    const float bq  = b_qkv[n0 + n];
    const float scl = (s == 0) ? 0.17677669529663687f : 1.0f;  // H^-0.5
    unsigned short* dst = (s == 0) ? Qg : ((s == 1) ? Kg : Vg);
#pragma unroll
    for (int mt = 0; mt < 8; ++mt) {
#pragma unroll
      for (int r = 0; r < 8; ++r) {
        size_t m  = mbase + (mt << 4) + r + hiM;   // global token row
        size_t bw = m / 49;
        size_t c  = m - bw * 49;
        float v = (acc[mt][r] + bq) * scl;
        dst[((bw * 32 + h) * 49 + c) * 16 + n] = f2bf(v);
      }
    }
  }
}

// ---------------------------------------------------------------------------
// K2: attention.  One WG per (window b, group of 8 heads); one wave per head.
// Q/K staged [c][16] (A and B fragments both K-contiguous), V staged
// transposed [d][64] for the P@V B-operand.  Softmax over 64 padded keys via
// shfl_xor reductions inside the 16-lane halves (wave32).  P bounced through
// LDS (stride 72 halfs, conflict-free, 16B aligned) to become A-fragments.
// ---------------------------------------------------------------------------
__global__ __launch_bounds__(256) void k2_attn(const unsigned short* __restrict__ Qg,
                                               const unsigned short* __restrict__ Kg,
                                               const unsigned short* __restrict__ Vg,
                                               const unsigned char* __restrict__ mask,
                                               const float* __restrict__ biasPad,
                                               unsigned short* __restrict__ AOg) {
  __shared__ __align__(16) unsigned short Qs [8 * 64 * 24];
  __shared__ __align__(16) unsigned short Ks [8 * 64 * 24];
  __shared__ __align__(16) unsigned short Vts[8 * 16 * 72];
  __shared__ __align__(16) unsigned short Ps [8 * 64 * 72];
  __shared__ float maskadd[64];

  const int tid = threadIdx.x;
  const int b  = blockIdx.x >> 2;
  const int hg = blockIdx.x & 3;

  if (tid < 64)
    maskadd[tid] = (tid < 49 && mask[(size_t)b * 49 * 512 + (size_t)tid * 512] != 0)
                       ? 0.0f : -1e30f;            // key mask = mask[b, key, 0]

  for (int i = tid; i < 8192; i += 256) {          // Q,K: 8 heads x 64 x 16
    int w = i >> 10, c = (i >> 4) & 63, d = i & 15;
    unsigned short qv = 0, kv = 0;
    if (c < 49) {
      size_t o = (((size_t)b * 32 + hg * 8 + w) * 49 + c) * 16 + d;
      qv = Qg[o]; kv = Kg[o];
    }
    Qs[w * 1536 + c * 24 + d] = qv;
    Ks[w * 1536 + c * 24 + d] = kv;
  }
  for (int i = tid; i < 8192; i += 256) {          // V transposed: [d][64]
    int w = i >> 10, d = (i >> 6) & 15, c = i & 63;
    unsigned short vv = 0;
    if (c < 49)
      vv = Vg[(((size_t)b * 32 + hg * 8 + w) * 49 + c) * 16 + d];
    Vts[w * 1152 + d * 72 + c] = vv;
  }
  __syncthreads();

  const int wave = tid >> 5, lane = tid & 31;
  const int h = hg * 8 + wave;
  const int n = lane & 15, hiM = (lane >> 4) << 3;
  const unsigned short* Qh  = Qs  + wave * 1536;
  const unsigned short* Kh  = Ks  + wave * 1536;
  const unsigned short* Vth = Vts + wave * 1152;
  unsigned short* P = Ps + wave * 4608;
  const float* biasH = biasPad + (h << 12);
  const v8f vzero = {0.f,0.f,0.f,0.f,0.f,0.f,0.f,0.f};

  v16bf kb[4];
#pragma unroll
  for (int nt = 0; nt < 4; ++nt) kb[nt] = load_frag_k16(Kh, nt << 4, 24, lane);

  for (int mt = 0; mt < 4; ++mt) {
    const v16bf qa = load_frag_k16(Qh, mt << 4, 24, lane);
    v8f s[4];
#pragma unroll
    for (int nt = 0; nt < 4; ++nt)
      s[nt] = __builtin_amdgcn_wmma_f32_16x16x32_bf16(
          false, qa, false, kb[nt], (short)0, vzero, false, false);

#pragma unroll
    for (int nt = 0; nt < 4; ++nt) {               // + rel bias + key mask
      const int nn = (nt << 4) + n;
      const float ma = maskadd[nn];
#pragma unroll
      for (int r = 0; r < 8; ++r) {
        const int m = (mt << 4) + r + hiM;
        s[nt][r] += biasH[(m << 6) + nn] + ma;
      }
    }
#pragma unroll
    for (int r = 0; r < 8; ++r) {                  // softmax over 64 keys
      float mx = fmaxf(fmaxf(s[0][r], s[1][r]), fmaxf(s[2][r], s[3][r]));
      mx = fmaxf(mx, __shfl_xor(mx, 1, 32));
      mx = fmaxf(mx, __shfl_xor(mx, 2, 32));
      mx = fmaxf(mx, __shfl_xor(mx, 4, 32));
      mx = fmaxf(mx, __shfl_xor(mx, 8, 32));
      float sum = 0.0f;
#pragma unroll
      for (int nt = 0; nt < 4; ++nt) {
        float e = __expf(s[nt][r] - mx);
        s[nt][r] = e; sum += e;
      }
      sum += __shfl_xor(sum, 1, 32);
      sum += __shfl_xor(sum, 2, 32);
      sum += __shfl_xor(sum, 4, 32);
      sum += __shfl_xor(sum, 8, 32);
      const float inv = 1.0f / sum;
#pragma unroll
      for (int nt = 0; nt < 4; ++nt) s[nt][r] *= inv;
    }
#pragma unroll
    for (int nt = 0; nt < 4; ++nt)                 // P -> LDS (A layout bounce)
#pragma unroll
      for (int r = 0; r < 8; ++r)
        P[((mt << 4) + r + hiM) * 72 + (nt << 4) + n] = f2bf(s[nt][r]);
  }

  v8f oacc[4];
#pragma unroll
  for (int mt = 0; mt < 4; ++mt) oacc[mt] = vzero;
#pragma unroll
  for (int kt = 0; kt < 2; ++kt) {                 // P(64x64) @ V(64x16)
    const v16bf vb = load_frag(Vth, 0, 72, kt << 5, lane);
#pragma unroll
    for (int mt = 0; mt < 4; ++mt) {
      const v16bf pa = load_frag(P, mt << 4, 72, kt << 5, lane);
      oacc[mt] = __builtin_amdgcn_wmma_f32_16x16x32_bf16(
          false, pa, false, vb, (short)0, oacc[mt], false, false);
    }
  }
#pragma unroll
  for (int mt = 0; mt < 4; ++mt) {
#pragma unroll
    for (int r = 0; r < 8; ++r) {
      const int c = (mt << 4) + r + hiM;
      if (c < 49)
        AOg[((size_t)b * 49 + c) * 512 + (h << 4) + n] = f2bf(oacc[mt][r]);
    }
  }
}

// ---------------------------------------------------------------------------
// K3: out = AO @ Wproj + b_proj (fp32 out).  Same skeleton as K1, N=512.
// ---------------------------------------------------------------------------
__global__ __launch_bounds__(256) void k3_proj(const unsigned short* __restrict__ AOg,
                                               const float* __restrict__ b_proj,
                                               const unsigned short* __restrict__ wprojT,
                                               float* __restrict__ out) {
  __shared__ __align__(16) unsigned short xs[128 * 520];
  const int tid = threadIdx.x;
  const size_t mbase = (size_t)blockIdx.x * 128;

  for (int i = tid; i < 8192; i += 256) {          // stage AO (already bf16)
    int r = i >> 6, k8 = (i & 63) << 3;
    *(uint4*)(xs + r * 520 + k8) = *(const uint4*)(AOg + (mbase + r) * 512 + k8);
  }
  __syncthreads();

  const int wave = tid >> 5, lane = tid & 31;
  const int n = lane & 15, hiM = (lane >> 4) << 3;
  const v8f vzero = {0.f,0.f,0.f,0.f,0.f,0.f,0.f,0.f};

  for (int j = 0; j < 4; ++j) {                    // 32 n-tiles / 8 waves
    asm volatile("" ::: "memory");                 // keep LDS frag loads in-loop

    const int nt = wave + (j << 3);
    const int n0 = nt << 4;
    v8f acc[8];
#pragma unroll
    for (int mt = 0; mt < 8; ++mt) acc[mt] = vzero;

    for (int ks = 0; ks < 16; ++ks) {
      const v16bf bfrag = load_frag(wprojT, n0, 512, ks << 5, lane);
#pragma unroll
      for (int mt = 0; mt < 8; ++mt) {
        const v16bf afrag = load_frag(xs, mt << 4, 520, ks << 5, lane);
        acc[mt] = __builtin_amdgcn_wmma_f32_16x16x32_bf16(
            false, afrag, false, bfrag, (short)0, acc[mt], false, false);
      }
    }
    const float bp = b_proj[n0 + n];
#pragma unroll
    for (int mt = 0; mt < 8; ++mt)
#pragma unroll
      for (int r = 0; r < 8; ++r)
        out[(mbase + (mt << 4) + r + hiM) * 512 + n0 + n] = acc[mt][r] + bp;
  }
}

// ---------------------------------------------------------------------------
extern "C" void kernel_launch(void* const* d_in, const int* in_sizes, int n_in,
                              void* d_out, int out_size, void* d_ws, size_t ws_size,
                              hipStream_t stream) {
  (void)in_sizes; (void)n_in; (void)out_size; (void)ws_size;
  const float*         x     = (const float*)d_in[0];
  const unsigned char* mask  = (const unsigned char*)d_in[1];  // jnp bool = 1 byte
  const float*         Wqkv  = (const float*)d_in[2];
  const float*         bqkv  = (const float*)d_in[3];
  const float*         Wproj = (const float*)d_in[4];
  const float*         bproj = (const float*)d_in[5];
  const float*         btab  = (const float*)d_in[6];

  char* ws = (char*)d_ws;
  const size_t QKV_BYTES = (size_t)4096 * 32 * 49 * 16 * 2;    // 205,520,896
  unsigned short* wqkvT  = (unsigned short*)(ws);
  unsigned short* wprojT = (unsigned short*)(ws + 1572864);
  float*          biasPad= (float*)(ws + 2097152);
  unsigned short* Qg     = (unsigned short*)(ws + 2621440);
  unsigned short* Kg     = (unsigned short*)(ws + 2621440 + QKV_BYTES);
  unsigned short* Vg     = (unsigned short*)(ws + 2621440 + 2 * QKV_BYTES);
  unsigned short* AOg    = (unsigned short*)(ws + 2621440 + 3 * QKV_BYTES);

  k0_prep<<<4608, 256, 0, stream>>>(Wqkv, Wproj, btab, wqkvT, wprojT, biasPad);
  k1_qkv <<<1568, 256, 0, stream>>>(x, bqkv, wqkvT, Qg, Kg, Vg);
  k2_attn<<<16384, 256, 0, stream>>>(Qg, Kg, Vg, mask, biasPad, AOg);
  k3_proj<<<1568, 256, 0, stream>>>(AOg, bproj, wprojT, (float*)d_out);
}
